// ScriptableEncodeProcessDecode_57208964382820
// MI455X (gfx1250) — matrix operand order repaired and checked by reading, not compiled, not run on hardware
//
#include <hip/hip_runtime.h>
#include <hip/hip_bf16.h>
#include <stdint.h>

typedef __attribute__((ext_vector_type(16))) __bf16 v16bf;
typedef __attribute__((ext_vector_type(8)))  __bf16 v8bf;
typedef __attribute__((ext_vector_type(4)))  __bf16 v4bf;
typedef __attribute__((ext_vector_type(8)))  float  v8f;

#define LN_EPS 1e-5f

extern __shared__ __bf16 smem[];

// ---------------------------------------------------------------------------
// CDNA5 async global->LDS copy (ASYNCcnt-tracked), 16B per lane.
// ---------------------------------------------------------------------------
__device__ __forceinline__ void async_cp16(uint32_t lds_addr, uint64_t gaddr) {
  asm volatile("global_load_async_to_lds_b128 %0, %1, off"
               :: "v"(lds_addr), "v"(gaddr) : "memory");
}
__device__ __forceinline__ void async_wait() {
  asm volatile("s_wait_asynccnt 0x0" ::: "memory");
}

// ---------------------------------------------------------------------------
// WMMA operand loaders (16x16x32 bf16, wave32; A/B use this code's contiguous-K order)
// A (16x32, MxK): lane l: m=l&15, h=l>>4; 16 consecutive K from h*16
// B (32x16, KxN): lane l: n=l&15, ks=l>>4; 16 consecutive K from ks*16,
//                 weights stored transposed [N][Kp] bf16
// C/D (16x16 f32): lane l: n=l&15; VGPR j -> row m=(l>>4)*8+j
// ---------------------------------------------------------------------------
__device__ __forceinline__ v16bf cat88(v8bf lo, v8bf hi) {
  return __builtin_shufflevector(lo, hi, 0,1,2,3,4,5,6,7,8,9,10,11,12,13,14,15);
}

__device__ __forceinline__ v16bf load_A_lds(const __bf16* sA, int ldk, int kc) {
  int lane = threadIdx.x & 31;
  int m = lane & 15, h = lane >> 4;
  const __bf16* p = sA + m * ldk + kc * 32 + h * 16;
  v8bf lo = *(const v8bf*)p;
  v8bf hi = *(const v8bf*)(p + 8);
  return cat88(lo, hi);
}

__device__ __forceinline__ v16bf load_B_glb(const __bf16* Wt, int Kp, int t, int kc) {
  int lane = threadIdx.x & 31;
  int n = lane & 15, ks = lane >> 4;
  const __bf16* p = Wt + (size_t)(t * 16 + n) * Kp + kc * 32 + ks * 16;
  v8bf lo = *(const v8bf*)p;
  v8bf hi = *(const v8bf*)(p + 8);
  return cat88(lo, hi);
}

// MT row-tiles of 16 share each B tile; B software-pipelined one tile ahead.
template<int NT, int MT>
__device__ __forceinline__ void gemm_tiles(const __bf16* sA, int ldk,
                                           const __bf16* Wt, int Kp,
                                           v8f acc[MT][NT]) {
  #pragma unroll
  for (int m = 0; m < MT; ++m)
    #pragma unroll
    for (int t = 0; t < NT; ++t)
      #pragma unroll
      for (int j = 0; j < 8; ++j)
        acc[m][t][j] = 0.f;
  int nk = Kp >> 5;
  for (int kc = 0; kc < nk; ++kc) {
    v16bf a[MT];
    #pragma unroll
    for (int m = 0; m < MT; ++m)
      a[m] = load_A_lds(sA + m * 16 * ldk, ldk, kc);
    v16bf b = load_B_glb(Wt, Kp, 0, kc);
    #pragma unroll
    for (int t = 0; t < NT; ++t) {
      v16bf bn = b;
      if (t + 1 < NT) bn = load_B_glb(Wt, Kp, t + 1, kc);  // preload next tile
      #pragma unroll
      for (int m = 0; m < MT; ++m)
        acc[m][t] = __builtin_amdgcn_wmma_f32_16x16x32_bf16(
            false, a[m], false, b, (short)0, acc[m][t], false, false);
      b = bn;
    }
  }
}

template<int NT, int MT>
__device__ __forceinline__ void bias_act(v8f acc[MT][NT], const float* b, int nb,
                                         bool relu) {
  int n = threadIdx.x & 15;
  #pragma unroll
  for (int m = 0; m < MT; ++m)
    #pragma unroll
    for (int t = 0; t < NT; ++t) {
      int col = t * 16 + n;
      float bv = (col < nb) ? b[col] : 0.f;
      #pragma unroll
      for (int j = 0; j < 8; ++j) {
        float v = acc[m][t][j] + bv;
        acc[m][t][j] = relu ? fmaxf(v, 0.f) : v;
      }
    }
}

template<int MT>
__device__ __forceinline__ void act_to_lds(__bf16* sA, int ldk,
                                           const v8f acc[MT][8]) {
  int lane = threadIdx.x & 31;
  int n = lane & 15, mh = (lane >> 4) * 8;
  #pragma unroll
  for (int m = 0; m < MT; ++m)
    #pragma unroll
    for (int t = 0; t < 8; ++t)
      #pragma unroll
      for (int j = 0; j < 8; ++j)
        sA[(m * 16 + mh + j) * ldk + t * 16 + n] = (__bf16)acc[m][t][j];
}

// 3-layer MLP: K1p -> 128 (ReLU) -> 128 (ReLU) -> NT3*16 (linear)
template<int NT3, int MT>
__device__ __forceinline__ void mlp3(__bf16* sA, int ldk, int K1p,
    const __bf16* W1, const float* b1,
    const __bf16* W2, const float* b2,
    const __bf16* W3, const float* b3, int nb3,
    v8f out[MT][NT3]) {
  v8f acc[MT][8];
  gemm_tiles<8, MT>(sA, ldk, W1, K1p, acc);
  bias_act<8, MT>(acc, b1, 128, true);
  act_to_lds<MT>(sA, ldk, acc);      // same-wave LDS: in-order, no barrier needed
  gemm_tiles<8, MT>(sA, ldk, W2, 128, acc);
  bias_act<8, MT>(acc, b2, 128, true);
  act_to_lds<MT>(sA, ldk, acc);
  gemm_tiles<NT3, MT>(sA, ldk, W3, 128, out);
  bias_act<NT3, MT>(out, b3, nb3, false);
}

// LayerNorm over 128 cols of one 16-row tile: per-lane partials over 8 N-tiles,
// then xor-shuffle reduce across the 16-lane half that holds the same rows.
__device__ __forceinline__ void layer_norm8(v8f acc[8], const float* g, const float* bt) {
  int lane = threadIdx.x & 31;
  int n = lane & 15;
  #pragma unroll
  for (int j = 0; j < 8; ++j) {
    float s = 0.f, q = 0.f;
    #pragma unroll
    for (int t = 0; t < 8; ++t) { float v = acc[t][j]; s += v; q += v * v; }
    #pragma unroll
    for (int m = 1; m < 16; m <<= 1) {
      s += __shfl_xor(s, m, 32);
      q += __shfl_xor(q, m, 32);
    }
    float mu = s * (1.f / 128.f);
    float var = q * (1.f / 128.f) - mu * mu;
    float r = rsqrtf(var + LN_EPS);
    #pragma unroll
    for (int t = 0; t < 8; ++t) {
      int col = t * 16 + n;
      acc[t][j] = (acc[t][j] - mu) * r * g[col] + bt[col];
    }
  }
}

__device__ __forceinline__ v4bf to_bf4(float4 v) {
  v4bf r; r[0] = (__bf16)v.x; r[1] = (__bf16)v.y; r[2] = (__bf16)v.z; r[3] = (__bf16)v.w;
  return r;
}

// ---------------------------------------------------------------------------
// Weight repack: f32 [K,N] row-major -> bf16 [Np][Kp] (transposed, zero-pad)
// ---------------------------------------------------------------------------
__global__ void __launch_bounds__(256)
pack_w_kernel(const float* __restrict__ src, __bf16* __restrict__ dst,
              int K, int N, int Kp, int Np) {
  int i = blockIdx.x * blockDim.x + threadIdx.x;
  if (i >= Kp * Np) return;
  int n = i / Kp, k = i - n * Kp;
  float v = (k < K && n < N) ? src[(size_t)k * N + n] : 0.f;
  dst[i] = (__bf16)v;
}

// ---------------------------------------------------------------------------
// Generic row MLP. INBF: input rows are bf16 (latent state). OUTBF: write bf16.
// 32 rows per wave, ldk = 128, 4 waves / block.
// ---------------------------------------------------------------------------
template<int NT3, bool INBF, bool OUTBF>
__global__ void __launch_bounds__(128)
mlp_rows_kernel(const void* __restrict__ inv, int in_dim, int K1p,
    const __bf16* W1, const float* b1, const __bf16* W2, const float* b2,
    const __bf16* W3, const float* b3, int nb3,
    const float* ln_g, const float* ln_b,
    void* __restrict__ outv, int out_dim, int rows) {
  int wave = threadIdx.x >> 5, lane = threadIdx.x & 31;
  int tile = blockIdx.x * (blockDim.x >> 5) + wave;
  int row0 = tile * 32;
  if (row0 >= rows) return;
  __bf16* sA = smem + (size_t)wave * 32 * 128;
  for (int m = 0; m < 32; ++m) {
    for (int c = lane; c < K1p; c += 32) {
      __bf16 v;
      if (c >= in_dim) v = (__bf16)0.f;
      else if constexpr (INBF) v = ((const __bf16*)inv)[(size_t)(row0 + m) * in_dim + c];
      else v = (__bf16)((const float*)inv)[(size_t)(row0 + m) * in_dim + c];
      sA[m * 128 + c] = v;
    }
  }
  v8f acc[2][NT3];
  mlp3<NT3, 2>(sA, 128, K1p, W1, b1, W2, b2, W3, b3, nb3, acc);
  if constexpr (NT3 == 8) {
    if (ln_g) {
      layer_norm8(acc[0], ln_g, ln_b);
      layer_norm8(acc[1], ln_g, ln_b);
    }
  }
  int n = lane & 15, mh = (lane >> 4) * 8;
  #pragma unroll
  for (int mt = 0; mt < 2; ++mt)
    #pragma unroll
    for (int t = 0; t < NT3; ++t) {
      int col = t * 16 + n;
      if (col < out_dim)
        #pragma unroll
        for (int j = 0; j < 8; ++j) {
          size_t idx = (size_t)(row0 + mt * 16 + mh + j) * out_dim + col;
          if constexpr (OUTBF) ((__bf16*)outv)[idx] = (__bf16)acc[mt][t][j];
          else                 ((float*)outv)[idx]  = acc[mt][t][j];
        }
    }
}

// ---------------------------------------------------------------------------
// Edge message: msg = LN(MLP3([x[dst], x[src], e])); e += msg (bf16 state);
// agg[dst] += e_new (f32 atomics). 32 edges/wave, ldk=384, 2 waves/block.
// Staging is pure async global->LDS (bf16 rows = 256B = 16 lanes x 16B).
// ---------------------------------------------------------------------------
__global__ void __launch_bounds__(64)
edge_msg_kernel(const __bf16* __restrict__ x, __bf16* __restrict__ e,
    const int* __restrict__ eidx,
    const __bf16* W1, const float* b1, const __bf16* W2, const float* b2,
    const __bf16* W3, const float* b3,
    const float* ln_g, const float* ln_b,
    float* __restrict__ agg, int E) {
  const int LDK = 384;                       // elements; 768 bytes per row
  int wave = threadIdx.x >> 5, lane = threadIdx.x & 31;
  int tile = blockIdx.x * (blockDim.x >> 5) + wave;
  int e0 = tile * 32;
  if (e0 >= E) return;
  __bf16* sA = smem + (size_t)wave * 32 * LDK;
  __shared__ int sDstAll[2 * 32];
  int* sDst = sDstAll + wave * 32;
  const int* srcI = eidx;
  const int* dstI = eidx + E;
  sDst[lane] = dstI[e0 + lane];
  uint32_t sAu = (uint32_t)(uintptr_t)sA;
  int half = lane >> 4, li = lane & 15;
  // x[dst] -> bytes [0,256), x[src] -> [256,512): one row per iter, halves split.
  for (int m = 0; m < 32; ++m) {
    int eid = e0 + m;
    int node = half ? srcI[eid] : dstI[eid];
    uint64_t g = (uint64_t)(uintptr_t)(x + (size_t)node * 128) + (uint64_t)li * 16;
    async_cp16(sAu + (uint32_t)(m * LDK * 2 + half * 256 + li * 16), g);
  }
  // e rows -> bytes [512,768): two rows per iter (one per half-wave).
  for (int mm = 0; mm < 16; ++mm) {
    int m = mm * 2 + half;
    uint64_t g = (uint64_t)(uintptr_t)(e + (size_t)(e0 + m) * 128) + (uint64_t)li * 16;
    async_cp16(sAu + (uint32_t)(m * LDK * 2 + 512 + li * 16), g);
  }
  async_wait();
  v8f acc[2][8];
  mlp3<8, 2>(sA, LDK, LDK, W1, b1, W2, b2, W3, b3, 128, acc);
  layer_norm8(acc[0], ln_g, ln_b);
  layer_norm8(acc[1], ln_g, ln_b);
  int n = lane & 15, mh = (lane >> 4) * 8;
  #pragma unroll
  for (int mt = 0; mt < 2; ++mt)
    #pragma unroll
    for (int t = 0; t < 8; ++t) {
      int col = t * 16 + n;
      #pragma unroll
      for (int j = 0; j < 8; ++j) {
        int m = mt * 16 + mh + j;
        float vnew = (float)sA[m * LDK + 256 + col] + acc[mt][t][j]; // e seg intact
        e[(size_t)(e0 + m) * 128 + col] = (__bf16)vnew;
        unsafeAtomicAdd(agg + (size_t)sDst[m] * 128 + col, vnew);
      }
    }
}

// ---------------------------------------------------------------------------
// Node update: x += LN(MLP3([x, agg_b, agg_c, agg_t])). 32 rows/wave, ldk=512,
// 1 wave per block. x segment staged async (bf16); agg segments f32->bf16.
// ---------------------------------------------------------------------------
__global__ void __launch_bounds__(32)
node_update_kernel(__bf16* __restrict__ x,
    const float* __restrict__ gb, const float* __restrict__ gc,
    const float* __restrict__ gt,
    const __bf16* W1, const float* b1, const __bf16* W2, const float* b2,
    const __bf16* W3, const float* b3,
    const float* ln_g, const float* ln_b, int rows) {
  const int LDK = 512;                       // elements; 1024 bytes per row
  int lane = threadIdx.x & 31;
  int r0 = blockIdx.x * 32;
  if (r0 >= rows) return;
  __bf16* sA = smem;
  uint32_t sAu = (uint32_t)(uintptr_t)sA;
  int half = lane >> 4, li = lane & 15;
  for (int mm = 0; mm < 16; ++mm) {          // x rows, 2 per iter
    int m = mm * 2 + half;
    uint64_t g = (uint64_t)(uintptr_t)(x + (size_t)(r0 + m) * 128) + (uint64_t)li * 16;
    async_cp16(sAu + (uint32_t)(m * LDK * 2 + li * 16), g);
  }
  int c4 = lane * 4;
  for (int m = 0; m < 32; ++m) {
    size_t ro = (size_t)(r0 + m) * 128;
    *(v4bf*)(sA + m * LDK + 128 + c4) = to_bf4(*(const float4*)(gb + ro + c4));
    *(v4bf*)(sA + m * LDK + 256 + c4) = to_bf4(*(const float4*)(gc + ro + c4));
    *(v4bf*)(sA + m * LDK + 384 + c4) = to_bf4(*(const float4*)(gt + ro + c4));
  }
  async_wait();
  v8f acc[2][8];
  mlp3<8, 2>(sA, LDK, LDK, W1, b1, W2, b2, W3, b3, 128, acc);
  layer_norm8(acc[0], ln_g, ln_b);
  layer_norm8(acc[1], ln_g, ln_b);
  int n = lane & 15, mh = (lane >> 4) * 8;
  #pragma unroll
  for (int mt = 0; mt < 2; ++mt)
    #pragma unroll
    for (int t = 0; t < 8; ++t)
      #pragma unroll
      for (int j = 0; j < 8; ++j) {
        size_t idx = (size_t)(r0 + mt * 16 + mh + j) * 128 + t * 16 + n;
        x[idx] = (__bf16)((float)x[idx] + acc[mt][t][j]);
      }
}

// ---------------------------------------------------------------------------
// Host side
// ---------------------------------------------------------------------------
extern "C" void kernel_launch(void* const* d_in, const int* in_sizes, int n_in,
                              void* d_out, int out_size, void* d_ws, size_t ws_size,
                              hipStream_t stream) {
  (void)in_sizes; (void)n_in; (void)out_size; (void)ws_size;
  constexpr int NN = 20000, EB = 160000, EC = 40000, ET = 80000;
  const float* node_x = (const float*)d_in[0];
  const float* bea = (const float*)d_in[1];
  const float* cea = (const float*)d_in[2];
  const float* tea = (const float*)d_in[3];
  const int* bei = (const int*)d_in[4];
  const int* cei = (const int*)d_in[5];
  const int* tei = (const int*)d_in[6];
  const int P = 7;  // first param leaf; pytree dicts flatten in sorted-key order

  char* wsb = (char*)d_ws;
  __bf16* x  = (__bf16*)wsb; wsb += (size_t)NN * 128 * 2;
  __bf16* be = (__bf16*)wsb; wsb += (size_t)EB * 128 * 2;
  __bf16* ce = (__bf16*)wsb; wsb += (size_t)EC * 128 * 2;
  __bf16* te = (__bf16*)wsb; wsb += (size_t)ET * 128 * 2;
  float* aggb = (float*)wsb; wsb += (size_t)NN * 128 * 4;
  float* aggc = (float*)wsb; wsb += (size_t)NN * 128 * 4;
  float* aggt = (float*)wsb; wsb += (size_t)NN * 128 * 4;
  __bf16* wpk = (__bf16*)wsb;
  size_t pk = 0;

  auto packW = [&](int leaf, int K, int N, int Kp, int Np) -> const __bf16* {
    __bf16* dst = wpk + pk; pk += (size_t)Kp * Np;
    int tot = Kp * Np;
    pack_w_kernel<<<dim3((tot + 255) / 256), dim3(256), 0, stream>>>(
        (const float*)d_in[leaf], dst, K, N, Kp, Np);
    return dst;
  };

  struct BW { const __bf16 *w1, *w2, *w3; const float *b1, *b2, *b3, *g, *bn; };
  auto packBlock = [&](int base, int K1, int K1p) -> BW {
    // leaf order in a block: ln.b, ln.g, mlp0.b, mlp0.w, mlp1.b, mlp1.w, mlp2.b, mlp2.w
    BW r;
    r.bn = (const float*)d_in[base];
    r.g  = (const float*)d_in[base + 1];
    r.b1 = (const float*)d_in[base + 2]; r.w1 = packW(base + 3, K1, 128, K1p, 128);
    r.b2 = (const float*)d_in[base + 4]; r.w2 = packW(base + 5, 128, 128, 128, 128);
    r.b3 = (const float*)d_in[base + 6]; r.w3 = packW(base + 7, 128, 128, 128, 128);
    return r;
  };

  // params sorted keys: dec (6 leaves), enc_body, enc_cable, enc_contact, enc_node, proc
  BW encB = packBlock(P + 6 + 0,  4, 32);
  BW encC = packBlock(P + 6 + 8,  4, 32);
  BW encT = packBlock(P + 6 + 16, 4, 32);
  BW encN = packBlock(P + 6 + 24, 30, 32);
  const __bf16* dw1 = packW(P + 1, 128, 128, 128, 128);
  const __bf16* dw2 = packW(P + 3, 128, 128, 128, 128);
  const __bf16* dw3 = packW(P + 5, 128, 3, 128, 16);
  const float* db1 = (const float*)d_in[P + 0];
  const float* db2 = (const float*)d_in[P + 2];
  const float* db3 = (const float*)d_in[P + 4];
  const int Q = P + 38;  // proc: 4 steps x {body,cable,contact,update} x 8 leaves
  BW proc[4][4];
  for (int s = 0; s < 4; ++s)
    for (int k = 0; k < 4; ++k)
      proc[s][k] = packBlock(Q + s * 32 + k * 8, k < 3 ? 384 : 512, k < 3 ? 384 : 512);

  auto enc_launch = [&](const float* in, int in_dim, const BW& W,
                        __bf16* out, int rows) {
    int tiles = (rows + 31) / 32, wpb = 4;
    mlp_rows_kernel<8, false, true>
        <<<dim3((tiles + wpb - 1) / wpb), dim3(32 * wpb),
           (size_t)wpb * 32 * 128 * 2, stream>>>(
        in, in_dim, 32, W.w1, W.b1, W.w2, W.b2, W.w3, W.b3, 128,
        W.g, W.bn, out, 128, rows);
  };
  enc_launch(node_x, 30, encN, x, NN);
  enc_launch(bea, 4, encB, be, EB);
  enc_launch(cea, 4, encC, ce, EC);
  enc_launch(tea, 4, encT, te, ET);

  for (int s = 0; s < 4; ++s) {
    hipMemsetAsync(aggb, 0, (size_t)3 * NN * 128 * sizeof(float), stream);
    auto edge_launch = [&](__bf16* ebuf, const int* idx, int E, const BW& W,
                           float* agg) {
      int tiles = E / 32, wpb = 2;
      edge_msg_kernel<<<dim3(tiles / wpb), dim3(32 * wpb),
                        (size_t)wpb * 32 * 384 * 2, stream>>>(
          x, ebuf, idx, W.w1, W.b1, W.w2, W.b2, W.w3, W.b3, W.g, W.bn, agg, E);
    };
    edge_launch(be, bei, EB, proc[s][0], aggb);
    edge_launch(ce, cei, EC, proc[s][1], aggc);
    edge_launch(te, tei, ET, proc[s][2], aggt);
    {
      const BW& W = proc[s][3];
      node_update_kernel<<<dim3(NN / 32), dim3(32),
                           (size_t)32 * 512 * 2, stream>>>(
          x, aggb, aggc, aggt, W.w1, W.b1, W.w2, W.b2, W.w3, W.b3, W.g, W.bn, NN);
    }
  }

  {  // decoder: 128 -> 128 -> 128 -> 3, bf16 in, f32 out, no LN
    int tiles = NN / 32, wpb = 4;
    mlp_rows_kernel<1, true, false>
        <<<dim3((tiles + wpb - 1) / wpb), dim3(32 * wpb),
           (size_t)wpb * 32 * 128 * 2, stream>>>(
        x, 128, 128, dw1, db1, dw2, db2, dw3, db3, 3,
        (const float*)nullptr, (const float*)nullptr, d_out, 3, NN);
  }
}
